// BiLSTM_CRF_41326175322113
// MI455X (gfx1250) — matrix operand order repaired and checked by reading, not compiled, not run on hardware
//
#include <hip/hip_runtime.h>
#include <hip/hip_bf16.h>

#define BATCH 128
#define LSEQ  192
#define EMB   256
#define HDIM  256      // per-direction hidden
#define GATE4 1024     // 4*HDIM
#define NTAG  12
#define TSTART 10
#define TSTOP  11

typedef unsigned short u16;
typedef __bf16 bf16;
typedef bf16  v16bf __attribute__((ext_vector_type(16)));
typedef float v8f   __attribute__((ext_vector_type(8)));

union Frag { uint4 u[2]; v16bf v; };

__device__ __forceinline__ u16 f2bf(float f) {
  unsigned u = __float_as_uint(f);
  u += 0x7fffu + ((u >> 16) & 1u);           // round-to-nearest-even
  return (u16)(u >> 16);
}

__device__ __forceinline__ float sigm(float x) { return 1.0f / (1.0f + __expf(-x)); }

// ---------------- elementwise helpers ----------------

__global__ void f32_to_bf16_kernel(const float* __restrict__ src, u16* __restrict__ dst, int n) {
  int i = blockIdx.x * 256 + threadIdx.x;
  if (i < n) dst[i] = f2bf(src[i]);
}

__global__ void bias_combine_kernel(const float* __restrict__ bih, const float* __restrict__ bhh,
                                    float* __restrict__ out, int n) {
  int i = blockIdx.x * 256 + threadIdx.x;
  if (i < n) out[i] = bih[i] + bhh[i];
}

__global__ void zero_u32_kernel(unsigned* __restrict__ p, int n) {
  int i = blockIdx.x * blockDim.x + threadIdx.x;
  if (i < n) p[i] = 0u;
}

// Embedding gather: x = emb[sentence] -> bf16 (B*L, 256). 4 elems/thread.
__global__ void embed_gather_kernel(const float* __restrict__ emb, const int* __restrict__ sent,
                                    u16* __restrict__ xbf) {
  int g   = blockIdx.x * 256 + threadIdx.x;  // group of 4 elements
  int row = g >> 6;                          // b*L + t   (256/4 = 64 groups/row)
  int e4  = g & 63;
  int tok = sent[row];
  float4 v = *(const float4*)(emb + (size_t)tok * EMB + e4 * 4);
  ushort4 o;
  o.x = f2bf(v.x); o.y = f2bf(v.y); o.z = f2bf(v.z); o.w = f2bf(v.w);
  *(ushort4*)(xbf + (size_t)g * 4) = o;
}

// ---------------- WMMA GEMM:  C(MxN) = A(MxK,bf16) * W(NxK,bf16)^T + bias[n] ----------------
// grid = (N/64, M/128), block = 256 (8 waves). Wave w: 16 rows, 4 n-tiles.
// Double-buffered register pipeline: loads for chunk k+1 issue before WMMAs on chunk k.
// Requires K % 64 == 0 (K = 256 or 512 here).
__global__ void gemm_bf16_kernel(const u16* __restrict__ A, const u16* __restrict__ W,
                                 const float* __restrict__ bias, float* __restrict__ C,
                                 int N, int K) {
  const int wave = threadIdx.x >> 5;
  const int lane = threadIdx.x & 31;
  const int lo   = lane & 15, hi = lane >> 4;
  const int m0   = blockIdx.y * 128 + wave * 16;
  const int n0   = blockIdx.x * 64;

  const u16* arow = A + (size_t)(m0 + lo) * K + hi * 8;

  v8f acc[4];
  const v8f vz = {0.f,0.f,0.f,0.f,0.f,0.f,0.f,0.f};
  acc[0] = vz; acc[1] = vz; acc[2] = vz; acc[3] = vz;

  auto LD_A = [&](Frag& f, int kk) {
    const u16* p = arow + kk;
    f.u[0] = *(const uint4*)p;
    f.u[1] = *(const uint4*)(p + 16);
  };
  auto LD_B = [&](Frag* f, int kk) {
#pragma unroll
    for (int nt = 0; nt < 4; ++nt) {
      const u16* p = W + (size_t)(n0 + nt * 16 + lo) * K + hi * 8 + kk;
      f[nt].u[0] = *(const uint4*)p;
      f[nt].u[1] = *(const uint4*)(p + 16);
    }
  };

  Frag a0, a1, b0[4], b1[4];
  LD_A(a0, 0);
  LD_B(b0, 0);
  for (int kk = 0; kk < K; kk += 64) {
    LD_A(a1, kk + 32);
    LD_B(b1, kk + 32);
#pragma unroll
    for (int nt = 0; nt < 4; ++nt)
      acc[nt] = __builtin_amdgcn_wmma_f32_16x16x32_bf16(false, a0.v, false, b0[nt].v,
                                                        (short)0, acc[nt], false, false);
    if (kk + 64 < K) {
      LD_A(a0, kk + 64);
      LD_B(b0, kk + 64);
    }
#pragma unroll
    for (int nt = 0; nt < 4; ++nt)
      acc[nt] = __builtin_amdgcn_wmma_f32_16x16x32_bf16(false, a1.v, false, b1[nt].v,
                                                        (short)0, acc[nt], false, false);
  }
#pragma unroll
  for (int nt = 0; nt < 4; ++nt) {
    int n = n0 + nt * 16 + lo;
    float bb = bias[n];
#pragma unroll
    for (int v = 0; v < 8; ++v) {
      int m = m0 + hi * 8 + v;
      C[(size_t)m * N + n] = acc[nt][v] + bb;
    }
  }
}

// feats = h2(24576x512,bf16) @ fc_w(12x512,bf16)^T + fc_b   ->  fp32 (B*L,12)
__global__ void feats_gemm_kernel(const u16* __restrict__ A, const u16* __restrict__ W,
                                  const float* __restrict__ bias, float* __restrict__ C) {
  const int wave = threadIdx.x >> 5;
  const int lane = threadIdx.x & 31;
  const int lo   = lane & 15, hi = lane >> 4;
  const int m0   = blockIdx.x * 128 + wave * 16;

  v8f acc = {0.f,0.f,0.f,0.f,0.f,0.f,0.f,0.f};
  for (int kk = 0; kk < 512; kk += 32) {
    Frag a, b;
    const u16* ap = A + (size_t)(m0 + lo) * 512 + kk + hi * 8;
    a.u[0] = *(const uint4*)ap;
    a.u[1] = *(const uint4*)(ap + 16);
    if (lo < NTAG) {
      const u16* bp = W + (size_t)lo * 512 + kk + hi * 8;
      b.u[0] = *(const uint4*)bp;
      b.u[1] = *(const uint4*)(bp + 16);
    } else {
      uint4 z; z.x = 0; z.y = 0; z.z = 0; z.w = 0;
      b.u[0] = z; b.u[1] = z;
    }
    acc = __builtin_amdgcn_wmma_f32_16x16x32_bf16(false, a.v, false, b.v,
                                                  (short)0, acc, false, false);
  }
  if (lo < NTAG) {
    float bb = bias[lo];
#pragma unroll
    for (int v = 0; v < 8; ++v)
      C[(size_t)(m0 + hi * 8 + v) * NTAG + lo] = acc[v] + bb;
  }
}

// ---------------- BiLSTM recurrence (one layer, both directions) ----------------
// grid = 32 WGs (dir = blk/16, h-slice = blk%16), block = 256 (8 waves).
// Wave w owns batch rows [16w,16w+16) x h-cols [16j,16j+16): c/h state in registers.
// Whh slice (4 gates x 16 rows x 256 K, bf16) cached in LDS. h ping-pongs in global.
// gx loads hoisted ahead of the recurrent WMMAs; next step's gx prefetched.
__global__ void lstm_layer_kernel(const float* __restrict__ gx,      // (B*L, 2048) fp32 (incl. biases)
                                  const u16*  __restrict__ whh,      // (2,1024,256) bf16, this layer
                                  u16*        __restrict__ hbuf,     // 4 x 128*256 bf16 (dir x pingpong)
                                  u16*        __restrict__ outbf,    // (B,L,512) bf16
                                  const int*  __restrict__ seqlen,
                                  unsigned*   __restrict__ bar) {
  const int dir  = blockIdx.x >> 4;
  const int wgj  = blockIdx.x & 15;
  const int wave = threadIdx.x >> 5;
  const int lane = threadIdx.x & 31;
  const int lo   = lane & 15, hi = lane >> 4;
  const int m0   = wave * 16;       // batch row base
  const int hc0  = wgj * 16;        // h column base

  __shared__ __align__(16) u16 ldsW[64 * 256];   // 32 KB
  {
    const u16* src = whh + (size_t)dir * GATE4 * HDIM;
    for (int it = threadIdx.x; it < 64 * 32; it += 256) {
      int lrow  = it >> 5;
      int chunk = it & 31;
      int gate  = lrow >> 4, r = lrow & 15;
      const u16* sp = src + (size_t)(gate * 256 + hc0 + r) * 256 + chunk * 8;
      *(uint4*)&ldsW[lrow * 256 + chunk * 8] = *(const uint4*)sp;
    }
  }
  __syncthreads();

  float c[8], hreg[8];
  int   blen[8];
#pragma unroll
  for (int v = 0; v < 8; ++v) {
    c[v] = 0.f; hreg[v] = 0.f;
    int b = m0 + hi * 8 + v;
    int sl = seqlen[b];
    blen[v] = sl < LSEQ ? sl : LSEQ;
  }

  unsigned* ctr = bar + dir * 64;   // separate cache lines per direction
  const v8f vz = {0.f,0.f,0.f,0.f,0.f,0.f,0.f,0.f};

  for (int step = 0; step < LSEQ; ++step) {
    const int t = dir ? (LSEQ - 1 - step) : step;

    // --- issue gx gate-input loads first: overlap global latency with WMMAs ---
    float gv[4][8];
#pragma unroll
    for (int v = 0; v < 8; ++v) {
      int b = m0 + hi * 8 + v;
      const float* gp = gx + (size_t)(b * LSEQ + t) * 2048 + dir * GATE4 + hc0 + lo;
      gv[0][v] = gp[0];
      gv[1][v] = gp[256];
      gv[2][v] = gp[512];
      gv[3][v] = gp[768];
      if (step + 1 < LSEQ) {                      // prefetch next step's gate inputs
        int tn = dir ? (t - 1) : (t + 1);
        const float* gq = gx + (size_t)(b * LSEQ + tn) * 2048 + dir * GATE4 + hc0 + lo;
        __builtin_prefetch(gq, 0, 3);
        __builtin_prefetch(gq + 512, 0, 3);
      }
    }

    v8f acc[4];
    acc[0] = vz; acc[1] = vz; acc[2] = vz; acc[3] = vz;

    if (step > 0) {
      const u16* hprev = hbuf + (size_t)((dir << 1) + ((step - 1) & 1)) * (BATCH * HDIM);
#pragma unroll
      for (int kk = 0; kk < 8; ++kk) {
        Frag a;
        const u16* ap = hprev + (size_t)(m0 + lo) * HDIM + kk * 32 + hi * 8;
        a.u[0] = *(const uint4*)ap;
        a.u[1] = *(const uint4*)(ap + 16);
#pragma unroll
        for (int g = 0; g < 4; ++g) {
          Frag b;
          const u16* bp = &ldsW[(g * 16 + lo) * 256 + kk * 32 + hi * 8];
          b.u[0] = *(const uint4*)bp;
          b.u[1] = *(const uint4*)(bp + 16);
          acc[g] = __builtin_amdgcn_wmma_f32_16x16x32_bf16(false, a.v, false, b.v,
                                                           (short)0, acc[g], false, false);
        }
      }
    }

    u16 hout[8];
#pragma unroll
    for (int v = 0; v < 8; ++v) {
      int b = m0 + hi * 8 + v;
      float xi = gv[0][v] + acc[0][v];
      float xf = gv[1][v] + acc[1][v];
      float xg = gv[2][v] + acc[2][v];
      float xo = gv[3][v] + acc[3][v];
      float ig = sigm(xi), fg = sigm(xf), gg = tanhf(xg), og = sigm(xo);
      float cn = fg * c[v] + ig * gg;
      float hn = og * tanhf(cn);
      bool m = (t < blen[v]);
      c[v]    = m ? cn : c[v];
      hreg[v] = m ? hn : hreg[v];
      float outv = m ? hreg[v] : 0.f;
      outbf[(size_t)(b * LSEQ + t) * 512 + dir * HDIM + hc0 + lo] = f2bf(outv);
      hout[v] = f2bf(hreg[v]);
    }
    u16* hcur = hbuf + (size_t)((dir << 1) + (step & 1)) * (BATCH * HDIM);
#pragma unroll
    for (int v = 0; v < 8; ++v) {
      int b = m0 + hi * 8 + v;
      hcur[(size_t)b * HDIM + hc0 + lo] = hout[v];
    }

    // device-scope step barrier (16 WGs per direction), monotonic counter
    __syncthreads();
    if (threadIdx.x == 0) {
      __threadfence();                     // release h writes device-wide
      atomicAdd(ctr, 1u);
      unsigned target = 16u * (unsigned)(step + 1);
      while (__hip_atomic_load(ctr, __ATOMIC_ACQUIRE, __HIP_MEMORY_SCOPE_AGENT) < target)
        __builtin_amdgcn_s_sleep(1);
    }
    __syncthreads();
    __builtin_amdgcn_fence(__ATOMIC_ACQUIRE, "agent");   // invalidate for fresh h reads
  }
}

// ---------------- CRF NLL: 1 WG, thread b handles batch b ----------------
__global__ void crf_kernel(const float* __restrict__ feats, const int* __restrict__ tags,
                           const float* __restrict__ trans, float* __restrict__ out) {
  __shared__ float tr[NTAG * NTAG];
  __shared__ float red[BATCH];
  for (int i = threadIdx.x; i < NTAG * NTAG; i += BATCH) tr[i] = trans[i];
  __syncthreads();

  const int b = threadIdx.x;
  float alpha[NTAG];
#pragma unroll
  for (int i = 0; i < NTAG; ++i) alpha[i] = -10000.0f;
  alpha[TSTART] = 0.0f;

  const float* fb = feats + (size_t)b * LSEQ * NTAG;
  for (int t = 0; t < LSEQ; ++t) {
    const float* ft = fb + t * NTAG;
    float na[NTAG];
#pragma unroll
    for (int i = 0; i < NTAG; ++i) {
      float mx = -3.4e38f;
#pragma unroll
      for (int j = 0; j < NTAG; ++j) { float v = alpha[j] + tr[i * NTAG + j]; mx = v > mx ? v : mx; }
      float s = 0.f;
#pragma unroll
      for (int j = 0; j < NTAG; ++j) s += __expf(alpha[j] + tr[i * NTAG + j] - mx);
      na[i] = mx + __logf(s) + ft[i];
    }
#pragma unroll
    for (int i = 0; i < NTAG; ++i) alpha[i] = na[i];
  }
  float mx = -3.4e38f;
#pragma unroll
  for (int i = 0; i < NTAG; ++i) { float v = alpha[i] + tr[TSTOP * NTAG + i]; mx = v > mx ? v : mx; }
  float s = 0.f;
#pragma unroll
  for (int i = 0; i < NTAG; ++i) s += __expf(alpha[i] + tr[TSTOP * NTAG + i] - mx);
  float logz = mx + __logf(s);

  int prev = TSTART;
  float gold = 0.f;
  for (int t = 0; t < LSEQ; ++t) {
    int cur = tags[b * LSEQ + t];
    gold += tr[cur * NTAG + prev] + fb[t * NTAG + cur];
    prev = cur;
  }
  gold += tr[TSTOP * NTAG + prev];

  red[b] = logz - gold;
  __syncthreads();
  for (int st = 64; st > 0; st >>= 1) {
    if (threadIdx.x < st) red[threadIdx.x] += red[threadIdx.x + st];
    __syncthreads();
  }
  if (threadIdx.x == 0) out[0] = red[0] / (float)BATCH;
}

// ---------------- workspace layout ----------------
static const size_t OFF_WIH0  = 0;                         // 2*1024*256 bf16 = 1,048,576 B
static const size_t OFF_WIH1  = OFF_WIH0 + 1048576;        // 2*1024*512 bf16 = 2,097,152 B
static const size_t OFF_WHH   = OFF_WIH1 + 2097152;        // 2*2*1024*256 bf16 = 2,097,152 B
static const size_t OFF_FCW   = OFF_WHH  + 2097152;        // 12*512 bf16 (pad 16K)
static const size_t OFF_BIAS  = OFF_FCW  + 16384;          // 4096 fp32 = 16,384 B
static const size_t OFF_XBF   = OFF_BIAS + 16384;          // 24576*256 bf16 = 12,582,912 B
static const size_t OFF_H1    = OFF_XBF  + 12582912;       // 24576*512 bf16 = 25,165,824 B
static const size_t OFF_H2    = OFF_H1   + 25165824;       // 25,165,824 B
static const size_t OFF_HB    = OFF_H2   + 25165824;       // 4 * 128*256 bf16 = 262,144 B
static const size_t OFF_BAR   = OFF_HB   + 262144;         // 128 u32 (pad 1K)
static const size_t OFF_FEATS = OFF_BAR  + 1024;           // 24576*12 fp32 = 1,179,648 B
static const size_t OFF_GX    = OFF_FEATS + 1179648;       // 24576*2048 fp32 = 201,326,592 B

extern "C" void kernel_launch(void* const* d_in, const int* in_sizes, int n_in,
                              void* d_out, int out_size, void* d_ws, size_t ws_size,
                              hipStream_t stream) {
  const float* emb    = (const float*)d_in[0];
  const float* w_ih0  = (const float*)d_in[1];
  const float* w_ih1  = (const float*)d_in[2];
  const float* w_hh   = (const float*)d_in[3];
  const float* b_ih   = (const float*)d_in[4];
  const float* b_hh   = (const float*)d_in[5];
  const float* fc_w   = (const float*)d_in[6];
  const float* fc_b   = (const float*)d_in[7];
  const float* trans  = (const float*)d_in[8];
  const int*   sent   = (const int*)d_in[9];
  const int*   seqlen = (const int*)d_in[10];
  const int*   tags   = (const int*)d_in[11];

  char* ws = (char*)d_ws;
  u16*      wih0  = (u16*)(ws + OFF_WIH0);
  u16*      wih1  = (u16*)(ws + OFF_WIH1);
  u16*      whh   = (u16*)(ws + OFF_WHH);
  u16*      fcw   = (u16*)(ws + OFF_FCW);
  float*    bias  = (float*)(ws + OFF_BIAS);
  u16*      xbf   = (u16*)(ws + OFF_XBF);
  u16*      h1    = (u16*)(ws + OFF_H1);
  u16*      h2    = (u16*)(ws + OFF_H2);
  u16*      hbuf  = (u16*)(ws + OFF_HB);
  unsigned* bar   = (unsigned*)(ws + OFF_BAR);
  float*    feats = (float*)(ws + OFF_FEATS);
  float*    gx    = (float*)(ws + OFF_GX);

  // weight/bias prep (bf16)
  f32_to_bf16_kernel<<<(524288 + 255) / 256, 256, 0, stream>>>(w_ih0, wih0, 524288);
  f32_to_bf16_kernel<<<(1048576 + 255) / 256, 256, 0, stream>>>(w_ih1, wih1, 1048576);
  f32_to_bf16_kernel<<<(1048576 + 255) / 256, 256, 0, stream>>>(w_hh, whh, 1048576);
  f32_to_bf16_kernel<<<(6144 + 255) / 256, 256, 0, stream>>>(fc_w, fcw, 6144);
  bias_combine_kernel<<<16, 256, 0, stream>>>(b_ih, b_hh, bias, 4096);

  // embedding gather -> bf16
  embed_gather_kernel<<<6144, 256, 0, stream>>>(emb, sent, xbf);

  // layer 1: input projection + recurrence
  gemm_bf16_kernel<<<dim3(32, 192), 256, 0, stream>>>(xbf, wih0, bias, gx, 2048, 256);
  zero_u32_kernel<<<1, 128, 0, stream>>>(bar, 128);
  lstm_layer_kernel<<<32, 256, 0, stream>>>(gx, whh, hbuf, h1, seqlen, bar);

  // layer 2
  gemm_bf16_kernel<<<dim3(32, 192), 256, 0, stream>>>(h1, wih1, bias + 2048, gx, 2048, 512);
  zero_u32_kernel<<<1, 128, 0, stream>>>(bar, 128);
  lstm_layer_kernel<<<32, 256, 0, stream>>>(gx, whh + 2 * 1024 * 256, hbuf, h2, seqlen, bar);

  // projection to tag space + CRF NLL
  feats_gemm_kernel<<<192, 256, 0, stream>>>(h2, fcw, fc_b, feats);
  crf_kernel<<<1, 128, 0, stream>>>(feats, tags, trans, (float*)d_out);
}
